// AutoFormerFT_86689619903555
// MI455X (gfx1250) — matrix operand order repaired and checked by reading, not compiled, not run on hardware
//
#include <hip/hip_runtime.h>
#include <hip/hip_bf16.h>
#include <math.h>

typedef __attribute__((ext_vector_type(16))) _Float16 v16h;
typedef __attribute__((ext_vector_type(8)))  float    v8f;
typedef _Float16 h16;

#define B_      32
#define L_      512
#define D_      768
#define DFF_    3072
#define M_      (B_ * L_)          // 16384 rows
#define NL_     2
#define TOPK_   6
#define MA_     25
#define CMID_   384
#define PLEN_   255
#define LATENT_ 32
#define NCELLS_ 2034
#define KCONV_  2304               // D*3

// ---------------------------------------------------------------------------
// Generic f32 -> f16 cast
// ---------------------------------------------------------------------------
__global__ __launch_bounds__(256)
void cast_f16_kernel(const float* __restrict__ s, h16* __restrict__ d, long n) {
  long i = (long)blockIdx.x * 256 + threadIdx.x;
  if (i < n) d[i] = (h16)s[i];
}

// ---------------------------------------------------------------------------
// WMMA GEMM:  C[m,n] = sum_k A[m,k] * W[n,k]  (+bias[n]), optional GELU,
// optional f32 and/or f16 outputs, optional batching via strides.
//
// Block tile 64(M) x 64(N), K-step 32. 8 waves as 2(M) x 4(N); each wave owns a
// 32x16 slab -> two v_wmma per K-step sharing one B fragment.
// Global->LDS staging uses CDNA5 GLOBAL_LOAD_ASYNC_TO_LDS_B128 with LDS
// double-buffering: tile i+1 streams in (ASYNCcnt) while tile i is multiplied.
// ---------------------------------------------------------------------------
#define BM 64
#define BN 64
#define BK 32

__global__ __launch_bounds__(256)
void gemm_wmma_kernel(const h16* __restrict__ A, const h16* __restrict__ W,
                      const float* __restrict__ bias,
                      float* __restrict__ C32, h16* __restrict__ C16,
                      int N, int K, int act,
                      long sA, long sW, long sC) {
  __shared__ h16 As[2][BM][BK];   // 2 x 4KB
  __shared__ h16 Ws[2][BN][BK];   // 2 x 4KB

  const int tid  = threadIdx.x;
  const int lane = tid & 31;
  const int w    = tid >> 5;
  const int wm   = w >> 2;     // 0..1 -> M slab of 32 rows
  const int wn   = w & 3;      // 0..3 -> N slab of 16 cols
  const int m0   = blockIdx.y * BM;
  const int n0   = blockIdx.x * BN;

  A += (long)blockIdx.z * sA;
  W += (long)blockIdx.z * sW;
  const long cbase = (long)blockIdx.z * sC;

  // tile-load mapping: 64x32 halves, 8 halves (16B) per thread
  const int t_row = (tid * 8) >> 5;   // 0..63
  const int t_k   = (tid * 8) & 31;   // 0,8,16,24

  // LDS destinations for async copies (generic ptr low 32 bits == LDS offset)
  const unsigned ldsA = (unsigned)(size_t)&As[0][t_row][t_k];
  const unsigned ldsW = (unsigned)(size_t)&Ws[0][t_row][t_k];
  const unsigned bufA = (unsigned)sizeof(As[0]);   // 4096B
  const unsigned bufW = (unsigned)sizeof(Ws[0]);   // 4096B

  const h16* gA = A + (long)(m0 + t_row) * K + t_k;
  const h16* gW = W + (long)(n0 + t_row) * K + t_k;

  auto issue = [&](int i) {
    const unsigned la = ldsA + (unsigned)(i & 1) * bufA;
    const unsigned lw = ldsW + (unsigned)(i & 1) * bufW;
    const h16* pa = gA + (long)i * BK;
    const h16* pw = gW + (long)i * BK;
    asm volatile("global_load_async_to_lds_b128 %0, %1, off\n\t"
                 "global_load_async_to_lds_b128 %2, %3, off"
                 :: "v"(la), "v"(pa), "v"(lw), "v"(pw)
                 : "memory");
  };

  const int r    = lane & 15;
  const int hsel = lane >> 4;
  const int nk   = K / BK;

  v8f acc0 = {};
  v8f acc1 = {};

  issue(0);
  for (int i = 0; i < nk; ++i) {
    if (i + 1 < nk) {
      issue(i + 1);
      asm volatile("s_wait_asynccnt 0x2" ::: "memory");  // tile i landed
    } else {
      asm volatile("s_wait_asynccnt 0x0" ::: "memory");
    }
    __syncthreads();   // all waves' tile-i data visible

    const h16 (*Ab)[BK] = As[i & 1];
    const h16 (*Wb)[BK] = Ws[i & 1];

    v16h af0, af1, bf;
#pragma unroll
    for (int j = 0; j < 8; ++j) {
      // A 16x32 f16 layout: lanes0-15 K:{0..7,16..23}, lanes16-31 K:{8..15,24..31}
      const int ka = (j < 4 ? 2 * j : 16 + 2 * (j - 4)) + 8 * hsel;
      af0[2 * j]     = Ab[wm * 32 + r][ka];
      af0[2 * j + 1] = Ab[wm * 32 + r][ka + 1];
      af1[2 * j]     = Ab[wm * 32 + 16 + r][ka];
      af1[2 * j + 1] = Ab[wm * 32 + 16 + r][ka + 1];
      // B 32x16 f16 layout: same K order as A: lanes0-15 K:{0..7,16..23}, lanes16-31 K:{8..15,24..31}
      const int kb = ka;
      bf[2 * j]     = Wb[wn * 16 + r][kb];
      bf[2 * j + 1] = Wb[wn * 16 + r][kb + 1];
    }
    acc0 = __builtin_amdgcn_wmma_f32_16x16x32_f16(false, af0, false, bf,
                                                  (short)0, acc0, false, false);
    acc1 = __builtin_amdgcn_wmma_f32_16x16x32_f16(false, af1, false, bf,
                                                  (short)0, acc1, false, false);
    __syncthreads();   // everyone done reading buf (i&1) before it is refilled
  }

  // C/D f32 16x16 layout: lanes0-15 rows 0..7, lanes16-31 rows 8..15, N=lane&15
  const int cn = n0 + wn * 16 + (lane & 15);
  const float bv = bias ? bias[cn] : 0.0f;
#pragma unroll
  for (int rr = 0; rr < 8; ++rr) {
    const int cm0 = m0 + wm * 32 + rr + 8 * hsel;
    float v0 = acc0[rr] + bv;
    float v1 = acc1[rr] + bv;
    if (act == 1) {
      v0 = 0.5f * v0 * (1.0f + erff(v0 * 0.70710678118f));  // exact GELU
      v1 = 0.5f * v1 * (1.0f + erff(v1 * 0.70710678118f));
    }
    const long o0 = cbase + (long)cm0 * N + cn;
    const long o1 = cbase + (long)(cm0 + 16) * N + cn;
    if (C32) { C32[o0] = v0; C32[o1] = v1; }
    if (C16) { C16[o0] = (h16)v0; C16[o1] = (h16)v1; }
  }
}

// ---------------------------------------------------------------------------
// Embedding + LayerNorm (eps 1e-12), writes x(f32) and x16(f16)
// ---------------------------------------------------------------------------
__global__ __launch_bounds__(256)
void embed_ln_kernel(const int* __restrict__ ids, const float* __restrict__ we,
                     const float* __restrict__ pe, const float* __restrict__ te,
                     const float* __restrict__ g, const float* __restrict__ bb,
                     float* __restrict__ x, h16* __restrict__ x16) {
  const int row = blockIdx.x;       // b*L + l
  const int l   = row & (L_ - 1);
  const long id = ids[row];
  __shared__ float red[256];

  float v[3], s = 0.0f;
#pragma unroll
  for (int i = 0; i < 3; ++i) {
    const int d = threadIdx.x + i * 256;
    v[i] = we[id * D_ + d] + pe[l * D_ + d] + te[d];
    s += v[i];
  }
  red[threadIdx.x] = s; __syncthreads();
  for (int o = 128; o > 0; o >>= 1) { if (threadIdx.x < o) red[threadIdx.x] += red[threadIdx.x + o]; __syncthreads(); }
  const float mu = red[0] / D_;
  __syncthreads();
  float s2 = 0.0f;
#pragma unroll
  for (int i = 0; i < 3; ++i) { const float t = v[i] - mu; s2 += t * t; }
  red[threadIdx.x] = s2; __syncthreads();
  for (int o = 128; o > 0; o >>= 1) { if (threadIdx.x < o) red[threadIdx.x] += red[threadIdx.x + o]; __syncthreads(); }
  const float rs = rsqrtf(red[0] / D_ + 1e-12f);
#pragma unroll
  for (int i = 0; i < 3; ++i) {
    const int d = threadIdx.x + i * 256;
    const float o = (v[i] - mu) * rs * g[d] + bb[d];
    x[(long)row * D_ + d] = o;
    x16[(long)row * D_ + d] = (h16)o;
  }
}

// ---------------------------------------------------------------------------
// Row LayerNorm over D (eps param), f32 in/out
// ---------------------------------------------------------------------------
__global__ __launch_bounds__(256)
void ln_rows_kernel(const float* __restrict__ in, float* __restrict__ out,
                    const float* __restrict__ g, const float* __restrict__ bb,
                    float eps) {
  const int row = blockIdx.x;
  __shared__ float red[256];
  float v[3], s = 0.0f;
#pragma unroll
  for (int i = 0; i < 3; ++i) { v[i] = in[(long)row * D_ + threadIdx.x + i * 256]; s += v[i]; }
  red[threadIdx.x] = s; __syncthreads();
  for (int o = 128; o > 0; o >>= 1) { if (threadIdx.x < o) red[threadIdx.x] += red[threadIdx.x + o]; __syncthreads(); }
  const float mu = red[0] / D_;
  __syncthreads();
  float s2 = 0.0f;
#pragma unroll
  for (int i = 0; i < 3; ++i) { const float t = v[i] - mu; s2 += t * t; }
  red[threadIdx.x] = s2; __syncthreads();
  for (int o = 128; o > 0; o >>= 1) { if (threadIdx.x < o) red[threadIdx.x] += red[threadIdx.x + o]; __syncthreads(); }
  const float rs = rsqrtf(red[0] / D_ + eps);
#pragma unroll
  for (int i = 0; i < 3; ++i) {
    const int d = threadIdx.x + i * 256;
    out[(long)row * D_ + d] = (v[i] - mu) * rs * g[d] + bb[d];
  }
}

// ---------------------------------------------------------------------------
// mean_value[b,tau] = (1/768) * sum_s Gram[b, (s+tau)%L, s]
// ---------------------------------------------------------------------------
__global__ __launch_bounds__(256)
void meandiag_kernel(const float* __restrict__ cmat, float* __restrict__ mv) {
  const int b = blockIdx.x >> 9;
  const int tau = blockIdx.x & 511;
  __shared__ float red[256];
  const float* cb = cmat + (long)b * L_ * L_;
  float s = 0.0f;
  for (int s0 = threadIdx.x; s0 < L_; s0 += 256) {
    const int t1 = (s0 + tau) & 511;
    s += cb[(long)t1 * L_ + s0];
  }
  red[threadIdx.x] = s; __syncthreads();
  for (int o = 128; o > 0; o >>= 1) { if (threadIdx.x < o) red[threadIdx.x] += red[threadIdx.x + o]; __syncthreads(); }
  if (threadIdx.x == 0) mv[(long)b * L_ + tau] = red[0] * (1.0f / 768.0f);
}

// ---------------------------------------------------------------------------
// per-batch top-6 over L=512 + softmax of the winning correlations
// ---------------------------------------------------------------------------
__global__ __launch_bounds__(256)
void topk_kernel(const float* __restrict__ mv, float* __restrict__ wout,
                 int* __restrict__ dout) {
  const int b = blockIdx.x, tid = threadIdx.x;
  __shared__ float sv[512];
  __shared__ float rv[256];
  __shared__ int   ri[256];
  __shared__ float wk[TOPK_];
  __shared__ int   dk[TOPK_];
  for (int i = tid; i < 512; i += 256) sv[i] = mv[(long)b * L_ + i];
  __syncthreads();
  for (int k = 0; k < TOPK_; ++k) {
    float best = sv[tid]; int bi = tid;
    const float o2 = sv[tid + 256];
    if (o2 > best) { best = o2; bi = tid + 256; }
    rv[tid] = best; ri[tid] = bi; __syncthreads();
    for (int o = 128; o > 0; o >>= 1) {
      if (tid < o && rv[tid + o] > rv[tid]) { rv[tid] = rv[tid + o]; ri[tid] = ri[tid + o]; }
      __syncthreads();
    }
    if (tid == 0) { wk[k] = rv[0]; dk[k] = ri[0]; sv[ri[0]] = -3.0e38f; }
    __syncthreads();
  }
  if (tid == 0) {
    float mx = wk[0];
    for (int k = 1; k < TOPK_; ++k) mx = fmaxf(mx, wk[k]);
    float se = 0.0f, e[TOPK_];
    for (int k = 0; k < TOPK_; ++k) { e[k] = expf(wk[k] - mx); se += e[k]; }
    for (int k = 0; k < TOPK_; ++k) { wout[b * TOPK_ + k] = e[k] / se; dout[b * TOPK_ + k] = dk[k]; }
  }
}

// ---------------------------------------------------------------------------
// delay aggregation: g16[b,l,d] = sum_k w[b,k] * v[b, (l+delay_k)%L, d]
// ---------------------------------------------------------------------------
__global__ __launch_bounds__(256)
void agg_kernel(const float* __restrict__ v, const float* __restrict__ w,
                const int* __restrict__ del, h16* __restrict__ g16) {
  const int row = blockIdx.x;    // b*L + l
  const int b = row >> 9, l = row & 511;
  __shared__ float ws[TOPK_];
  __shared__ int   ds[TOPK_];
  if (threadIdx.x < TOPK_) { ws[threadIdx.x] = w[b * TOPK_ + threadIdx.x]; ds[threadIdx.x] = del[b * TOPK_ + threadIdx.x]; }
  __syncthreads();
#pragma unroll
  for (int i = 0; i < 3; ++i) {
    const int d = threadIdx.x + i * 256;
    float s = 0.0f;
    for (int k = 0; k < TOPK_; ++k) {
      const int ll = (l + ds[k]) & 511;
      s += ws[k] * v[((long)b * L_ + ll) * D_ + d];
    }
    g16[(long)row * D_ + d] = (h16)s;
  }
}

// ---------------------------------------------------------------------------
// residual + series_decomp (moving average k=25, replicate pad), sliding window
// one thread per (b,d) column; writes f32 + f16 mirrors
// ---------------------------------------------------------------------------
__global__ __launch_bounds__(256)
void decomp_kernel(const float* __restrict__ x, const float* __restrict__ r,
                   float* __restrict__ xo, h16* __restrict__ xo16) {
  const int t = blockIdx.x * 256 + threadIdx.x;  // b*D + d
  const int b = t / D_, d = t % D_;
  const long base = (long)b * L_ * D_ + d;
  auto S = [&](int l) -> float {
    l = l < 0 ? 0 : (l > 511 ? 511 : l);
    const long o = base + (long)l * D_;
    return x[o] + r[o];
  };
  float wsum = 0.0f;
  for (int j = -12; j <= 12; ++j) wsum += S(j);
  for (int l = 0; l < L_; ++l) {
    const float o = S(l) - wsum * (1.0f / MA_);
    const long off = base + (long)l * D_;
    xo[off] = o;
    xo16[off] = (h16)o;
    wsum += S(l + 13) - S(l - 12);
  }
}

// ---------------------------------------------------------------------------
// att[b,l,d] = xh[b,l,d] - mean_l xh[b,:,d]
// ---------------------------------------------------------------------------
__global__ __launch_bounds__(256)
void colmean_sub_kernel(const float* __restrict__ in, float* __restrict__ out) {
  const int t = blockIdx.x * 256 + threadIdx.x;  // b*D + d
  const int b = t / D_, d = t % D_;
  const long base = (long)b * L_ * D_ + d;
  float s = 0.0f;
  for (int l = 0; l < L_; ++l) s += in[base + (long)l * D_];
  s *= (1.0f / L_);
  for (int l = 0; l < L_; ++l) out[base + (long)l * D_] = in[base + (long)l * D_] - s;
}

// ---------------------------------------------------------------------------
// im2col for conv1d(k=3, valid): ic[b*512+t, d*3+j] = att[b, t+j, d]; rows
// t>=510 zero-padded so the GEMM M stays 16384.
// ---------------------------------------------------------------------------
__global__ __launch_bounds__(256)
void im2col_kernel(const float* __restrict__ att, h16* __restrict__ ic) {
  const int row = blockIdx.x;    // b*512 + t
  const int b = row >> 9, t = row & 511;
  for (int c = threadIdx.x; c < KCONV_; c += 256) {
    float val = 0.0f;
    if (t < 510) {
      const int d = c / 3, j = c - 3 * d;
      val = att[((long)b * L_ + t + j) * D_ + d];
    }
    ic[(long)row * KCONV_ + c] = (h16)val;
  }
}

// ---------------------------------------------------------------------------
// BN + ELU + maxpool(2) + flatten:  f[b, o*255+u]
// conv out layout: co[(b*512+t)*384 + o]
// ---------------------------------------------------------------------------
__global__ __launch_bounds__(256)
void conv_pool_kernel(const float* __restrict__ co,
                      const float* __restrict__ bng, const float* __restrict__ bnb,
                      const float* __restrict__ bnm, const float* __restrict__ bnv,
                      float* __restrict__ f) {
  const int blk = blockIdx.x;
  const int b = blk / CMID_, o = blk % CMID_;
  const int u = threadIdx.x;
  if (u >= PLEN_) return;
  const float sc = rsqrtf(bnv[o] + 1e-5f) * bng[o];
  const float sh = bnb[o] - bnm[o] * sc;
  float a0 = co[((long)b * L_ + 2 * u) * CMID_ + o] * sc + sh;
  float a1 = co[((long)b * L_ + 2 * u + 1) * CMID_ + o] * sc + sh;
  a0 = a0 > 0.0f ? a0 : (expf(a0) - 1.0f);
  a1 = a1 > 0.0f ? a1 : (expf(a1) - 1.0f);
  f[(long)b * (CMID_ * PLEN_) + o * PLEN_ + u] = fmaxf(a0, a1);
}

// ---------------------------------------------------------------------------
// lin1: pre[b,i] = f[b,:] . lin1_w[i,:] + lin1_b[i]   (K = 97920)
// ---------------------------------------------------------------------------
__global__ __launch_bounds__(256)
void lin1_kernel(const float* __restrict__ f, const float* __restrict__ w,
                 const float* __restrict__ bias, float* __restrict__ pre) {
  const int i = blockIdx.x & 31, b = blockIdx.x >> 5;
  __shared__ float red[256];
  const float* fb = f + (long)b * (CMID_ * PLEN_);
  const float* wi = w + (long)i * (CMID_ * PLEN_);
  float s = 0.0f;
  for (int j = threadIdx.x; j < CMID_ * PLEN_; j += 256) s += fb[j] * wi[j];
  red[threadIdx.x] = s; __syncthreads();
  for (int o = 128; o > 0; o >>= 1) { if (threadIdx.x < o) red[threadIdx.x] += red[threadIdx.x + o]; __syncthreads(); }
  if (threadIdx.x == 0) pre[b * LATENT_ + i] = red[0] + bias[i];
}

// ---------------------------------------------------------------------------
// LN(32, eps 1e-5) + ELU -> h[b,i]
// ---------------------------------------------------------------------------
__global__ __launch_bounds__(32)
void lnelu_kernel(const float* __restrict__ pre, const float* __restrict__ g,
                  const float* __restrict__ bb, float* __restrict__ h) {
  const int b = blockIdx.x, i = threadIdx.x;
  __shared__ float sv[LATENT_];
  __shared__ float mu, rs;
  sv[i] = pre[b * LATENT_ + i];
  __syncthreads();
  if (i == 0) {
    float s = 0.0f;
    for (int j = 0; j < LATENT_; ++j) s += sv[j];
    mu = s / LATENT_;
    float s2 = 0.0f;
    for (int j = 0; j < LATENT_; ++j) { const float t = sv[j] - mu; s2 += t * t; }
    rs = rsqrtf(s2 / LATENT_ + 1e-5f);
  }
  __syncthreads();
  float o = (sv[i] - mu) * rs * g[i] + bb[i];
  o = o > 0.0f ? o : (expf(o) - 1.0f);
  h[b * LATENT_ + i] = o;
}

// ---------------------------------------------------------------------------
// lin2 + sigmoid -> out[b, c]
// ---------------------------------------------------------------------------
__global__ __launch_bounds__(256)
void final_kernel(const float* __restrict__ h, const float* __restrict__ w,
                  const float* __restrict__ bias, float* __restrict__ out) {
  const int b = blockIdx.y;
  const int c = blockIdx.x * 256 + threadIdx.x;
  __shared__ float hs[LATENT_];
  if (threadIdx.x < LATENT_) hs[threadIdx.x] = h[b * LATENT_ + threadIdx.x];
  __syncthreads();
  if (c < NCELLS_) {
    float s = bias[c];
#pragma unroll
    for (int i = 0; i < LATENT_; ++i) s += hs[i] * w[(long)c * LATENT_ + i];
    out[(long)b * NCELLS_ + c] = 1.0f / (1.0f + expf(-s));
  }
}

// ---------------------------------------------------------------------------
// Host orchestration
// ---------------------------------------------------------------------------
extern "C" void kernel_launch(void* const* d_in, const int* in_sizes, int n_in,
                              void* d_out, int out_size, void* d_ws, size_t ws_size,
                              hipStream_t stream) {
  (void)in_sizes; (void)n_in; (void)out_size; (void)ws_size;
  const int*   input_ids = (const int*)  d_in[0];
  const float* word_emb  = (const float*)d_in[1];
  const float* pos_emb   = (const float*)d_in[2];
  const float* tok_emb   = (const float*)d_in[3];
  const float* emb_g     = (const float*)d_in[4];
  const float* emb_b     = (const float*)d_in[5];
  const float* Wq = (const float*)d_in[6];  const float* bq = (const float*)d_in[7];
  const float* Wk = (const float*)d_in[8];  const float* bk = (const float*)d_in[9];
  const float* Wv = (const float*)d_in[10]; const float* bv = (const float*)d_in[11];
  const float* Wo = (const float*)d_in[12]; const float* bo = (const float*)d_in[13];
  const float* Wc1 = (const float*)d_in[14]; const float* bc1 = (const float*)d_in[15];
  const float* Wc2 = (const float*)d_in[16]; const float* bc2 = (const float*)d_in[17];
  const float* norm_g = (const float*)d_in[18]; const float* norm_b = (const float*)d_in[19];
  const float* conv_w = (const float*)d_in[20]; const float* conv_b = (const float*)d_in[21];
  const float* bn_g = (const float*)d_in[22]; const float* bn_b = (const float*)d_in[23];
  const float* bn_m = (const float*)d_in[24]; const float* bn_v = (const float*)d_in[25];
  const float* lin1_w = (const float*)d_in[26]; const float* lin1_b = (const float*)d_in[27];
  const float* ln2_g = (const float*)d_in[28]; const float* ln2_b = (const float*)d_in[29];
  const float* lin2_w = (const float*)d_in[30]; const float* lin2_b = (const float*)d_in[31];
  float* out = (float*)d_out;

  char* ws = (char*)d_ws;
  size_t o = 0;
  auto alloc = [&](size_t bytes) { size_t r = o; o += (bytes + 255) & ~(size_t)255; return r; };
  const size_t off_x    = alloc((size_t)M_ * D_ * 4);      // x f32
  const size_t off_x16  = alloc((size_t)M_ * D_ * 2);      // x f16
  const size_t off_q16  = alloc((size_t)M_ * D_ * 2);      // q f16
  const size_t off_k16  = alloc((size_t)M_ * D_ * 2);      // k f16
  const size_t off_v    = alloc((size_t)M_ * D_ * 4);      // v f32 / alt x / att
  const size_t off_g16  = alloc((size_t)M_ * D_ * 2);      // aggregated V f16
  const size_t off_a    = alloc((size_t)M_ * D_ * 4);      // attn/ffn residual f32 / xh
  const size_t off_tmp  = alloc((size_t)M_ * DFF_ * 2);    // ffn hidden f16 / im2col f16
  const size_t off_cmat = alloc((size_t)B_ * L_ * L_ * 4); // gram f32 / conv out f32
  const size_t off_f    = alloc((size_t)B_ * CMID_ * PLEN_ * 4);
  const size_t off_wq16 = alloc((size_t)NL_ * D_ * D_ * 2);
  const size_t off_wk16 = alloc((size_t)NL_ * D_ * D_ * 2);
  const size_t off_wv16 = alloc((size_t)NL_ * D_ * D_ * 2);
  const size_t off_wo16 = alloc((size_t)NL_ * D_ * D_ * 2);
  const size_t off_wc116 = alloc((size_t)NL_ * DFF_ * D_ * 2);
  const size_t off_wc216 = alloc((size_t)NL_ * D_ * DFF_ * 2);
  const size_t off_wcv16 = alloc((size_t)CMID_ * KCONV_ * 2);
  const size_t off_mv   = alloc((size_t)B_ * L_ * 4);
  const size_t off_wtop = alloc((size_t)B_ * TOPK_ * 4);
  const size_t off_dtop = alloc((size_t)B_ * TOPK_ * 4);
  const size_t off_pre  = alloc((size_t)B_ * LATENT_ * 4);
  const size_t off_h    = alloc((size_t)B_ * LATENT_ * 4);

  float* xA   = (float*)(ws + off_x);
  h16*   x16  = (h16*)(ws + off_x16);
  h16*   q16  = (h16*)(ws + off_q16);
  h16*   k16  = (h16*)(ws + off_k16);
  float* vbuf = (float*)(ws + off_v);     // v f32, then alt-x, then att
  h16*   g16  = (h16*)(ws + off_g16);
  float* abuf = (float*)(ws + off_a);
  h16*   t16  = (h16*)(ws + off_tmp);
  float* cmat = (float*)(ws + off_cmat);
  float* fbuf = (float*)(ws + off_f);
  float* mv   = (float*)(ws + off_mv);
  float* wtop = (float*)(ws + off_wtop);
  int*   dtop = (int*)(ws + off_dtop);
  float* pre  = (float*)(ws + off_pre);
  float* hbuf = (float*)(ws + off_h);

  auto cast = [&](const float* s, size_t dstOff, long n) {
    cast_f16_kernel<<<dim3((unsigned)((n + 255) / 256)), 256, 0, stream>>>(s, (h16*)(ws + dstOff), n);
  };
  // 1) weights -> f16 (conv_w [O,D,3] is already [384, 2304] row-major)
  cast(Wq, off_wq16, (long)NL_ * D_ * D_);
  cast(Wk, off_wk16, (long)NL_ * D_ * D_);
  cast(Wv, off_wv16, (long)NL_ * D_ * D_);
  cast(Wo, off_wo16, (long)NL_ * D_ * D_);
  cast(Wc1, off_wc116, (long)NL_ * DFF_ * D_);
  cast(Wc2, off_wc216, (long)NL_ * D_ * DFF_);
  cast(conv_w, off_wcv16, (long)CMID_ * KCONV_);

  // 2) embedding + LN
  embed_ln_kernel<<<M_, 256, 0, stream>>>(input_ids, word_emb, pos_emb, tok_emb,
                                          emb_g, emb_b, xA, x16);

  auto gemm = [&](const h16* A, const h16* W, const float* bias,
                  float* C32, h16* C16, int M, int N, int K, int act,
                  long sA, long sW, long sC, int batch) {
    gemm_wmma_kernel<<<dim3(N / BN, M / BM, batch), 256, 0, stream>>>(
        A, W, bias, C32, C16, N, K, act, sA, sW, sC);
  };

  // 3) encoder layers
  for (int l = 0; l < NL_; ++l) {
    const h16* wq = (h16*)(ws + off_wq16) + (long)l * D_ * D_;
    const h16* wk = (h16*)(ws + off_wk16) + (long)l * D_ * D_;
    const h16* wv = (h16*)(ws + off_wv16) + (long)l * D_ * D_;
    const h16* wo = (h16*)(ws + off_wo16) + (long)l * D_ * D_;
    const h16* wc1 = (h16*)(ws + off_wc116) + (long)l * DFF_ * D_;
    const h16* wc2 = (h16*)(ws + off_wc216) + (long)l * D_ * DFF_;

    // Q/K (f16 only), V (f32)
    gemm(x16, wq, bq + l * D_, nullptr, q16, M_, D_, D_, 0, 0, 0, 0, 1);
    gemm(x16, wk, bk + l * D_, nullptr, k16, M_, D_, D_, 0, 0, 0, 0, 1);
    gemm(x16, wv, bv + l * D_, vbuf, nullptr, M_, D_, D_, 0, 0, 0, 0, 1);

    // batched Gram: C[b] = q[b] (512x768) x k[b]^T -> circular autocorrelation
    gemm(q16, k16, nullptr, cmat, nullptr, L_, L_, D_, 0,
         (long)L_ * D_, (long)L_ * D_, (long)L_ * L_, B_);
    meandiag_kernel<<<B_ * L_, 256, 0, stream>>>(cmat, mv);
    topk_kernel<<<B_, 256, 0, stream>>>(mv, wtop, dtop);
    agg_kernel<<<M_, 256, 0, stream>>>(vbuf, wtop, dtop, g16);

    // output projection + residual + decomp (v buffer is dead -> alt x)
    gemm(g16, wo, bo + l * D_, abuf, nullptr, M_, D_, D_, 0, 0, 0, 0, 1);
    decomp_kernel<<<(B_ * D_) / 256, 256, 0, stream>>>(xA, abuf, vbuf, x16);

    // FFN: GELU fused into first GEMM, hidden kept f16-only
    gemm(x16, wc1, bc1 + l * DFF_, nullptr, t16, M_, DFF_, D_, 1, 0, 0, 0, 1);
    gemm(t16, wc2, bc2 + l * D_, abuf, nullptr, M_, D_, DFF_, 0, 0, 0, 0, 1);
    decomp_kernel<<<(B_ * D_) / 256, 256, 0, stream>>>(vbuf, abuf, xA, x16);
  }

  // 4) my_Layernorm + subtract per-(b,d) mean over L
  ln_rows_kernel<<<M_, 256, 0, stream>>>(xA, abuf, norm_g, norm_b, 1e-5f);
  colmean_sub_kernel<<<(B_ * D_) / 256, 256, 0, stream>>>(abuf, vbuf);  // att -> vbuf

  // 5) conv head as im2col + WMMA GEMM (M padded to 16384)
  im2col_kernel<<<M_, 256, 0, stream>>>(vbuf, t16);
  gemm(t16, (h16*)(ws + off_wcv16), conv_b, cmat, nullptr, M_, CMID_, KCONV_, 0, 0, 0, 0, 1);
  conv_pool_kernel<<<B_ * CMID_, 256, 0, stream>>>(cmat, bn_g, bn_b, bn_m, bn_v, fbuf);

  // 6) prediction head
  lin1_kernel<<<B_ * LATENT_, 256, 0, stream>>>(fbuf, lin1_w, lin1_b, pre);
  lnelu_kernel<<<B_, 32, 0, stream>>>(pre, ln2_g, ln2_b, hbuf);
  final_kernel<<<dim3((NCELLS_ + 255) / 256, B_), 256, 0, stream>>>(hbuf, lin2_w, lin2_b, out);
}
